// IntelligentRAGTransformer_34179349742188
// MI455X (gfx1250) — matrix operand order repaired and checked by reading, not compiled, not run on hardware
//
#include <hip/hip_runtime.h>
#include <math.h>

// ---- model dims (match reference) ----
#define Dm 1024
#define Tm 1024
#define Bm 2
#define Hm 16
#define Lm 8
#define Vm 32000
#define Mm (Bm * Tm)   // 2048 tokens

typedef __bf16 bf16;
typedef __bf16 v16bf __attribute__((ext_vector_type(16)));
typedef __bf16 v8bf  __attribute__((ext_vector_type(8)));
typedef float  v8f   __attribute__((ext_vector_type(8)));

union FragBF { v16bf bf; v8bf h[2]; bf16 s[16]; };

// A-matrix 16x32 bf16 fragment from LDS (row-major, row = M).
// Lanes 0-15 hold K 0..15, lanes 16-31 hold K 16..31.
__device__ __forceinline__ FragBF frag_a_lds(const bf16* row, int lane) {
  int kb = (lane < 16) ? 0 : 16;
  FragBF f;
  f.h[0] = *(const v8bf*)(row + kb);
  f.h[1] = *(const v8bf*)(row + kb + 8);
  return f;
}
// B-matrix 32x16 bf16 fragment from LDS stored N-major (row = N, 32 K elems).
// Lanes 0-15 hold K 0..15, lanes 16-31 hold K 16..31.
__device__ __forceinline__ FragBF frag_b_lds(const bf16* row, int lane) {
  int kb = (lane < 16) ? 0 : 16;
  FragBF f;
  f.h[0] = *(const v8bf*)(row + kb);
  f.h[1] = *(const v8bf*)(row + kb + 8);
  return f;
}
__device__ __forceinline__ v8f wmma_bf16(FragBF a, FragBF b, v8f c) {
  return __builtin_amdgcn_wmma_f32_16x16x32_bf16(
      false, a.bf, false, b.bf, (short)0, c, false, false);
}
__device__ __forceinline__ float gelu_exact(float x) {
  return 0.5f * x * (1.0f + erff(x * 0.70710678118654752f));
}
// convert 16 consecutive floats (4 x float4) to bf16 into dst[0..15]
__device__ __forceinline__ void cvt16(const float* __restrict__ src, bf16* dst) {
  float4 a0 = *(const float4*)(src);
  float4 a1 = *(const float4*)(src + 4);
  float4 a2 = *(const float4*)(src + 8);
  float4 a3 = *(const float4*)(src + 12);
  dst[0]  = (bf16)a0.x; dst[1]  = (bf16)a0.y; dst[2]  = (bf16)a0.z; dst[3]  = (bf16)a0.w;
  dst[4]  = (bf16)a1.x; dst[5]  = (bf16)a1.y; dst[6]  = (bf16)a1.z; dst[7]  = (bf16)a1.w;
  dst[8]  = (bf16)a2.x; dst[9]  = (bf16)a2.y; dst[10] = (bf16)a2.z; dst[11] = (bf16)a2.w;
  dst[12] = (bf16)a3.x; dst[13] = (bf16)a3.y; dst[14] = (bf16)a3.z; dst[15] = (bf16)a3.w;
}

// ---------------------------------------------------------------------------
// Embedding
// ---------------------------------------------------------------------------
__global__ __launch_bounds__(256)
void k_embed(const int* __restrict__ idx, const float* __restrict__ tok,
             const float* __restrict__ pos, float* __restrict__ x) {
  const int t  = blockIdx.x;
  const int tt = t & (Tm - 1);
  const int id = idx[t];
  const float* tr = tok + (size_t)id * Dm;
  const float* pr = pos + (size_t)tt * Dm;
  float* xr = x + (size_t)t * Dm;
  for (int i = threadIdx.x; i < Dm; i += 256) xr[i] = tr[i] + pr[i];
}

// ---------------------------------------------------------------------------
// LayerNorm (one token per block)
// ---------------------------------------------------------------------------
__global__ __launch_bounds__(256)
void k_layernorm(const float* __restrict__ x, const float* __restrict__ g,
                 const float* __restrict__ bta, float* __restrict__ out) {
  const int t = blockIdx.x;
  const float* xr = x + (size_t)t * Dm;
  __shared__ float rs[256], rq[256];
  float s = 0.f, q = 0.f;
  for (int i = threadIdx.x; i < Dm; i += 256) { float v = xr[i]; s += v; q += v * v; }
  rs[threadIdx.x] = s; rq[threadIdx.x] = q;
  __syncthreads();
  for (int off = 128; off > 0; off >>= 1) {
    if (threadIdx.x < off) { rs[threadIdx.x] += rs[threadIdx.x + off];
                             rq[threadIdx.x] += rq[threadIdx.x + off]; }
    __syncthreads();
  }
  const float mean = rs[0] * (1.0f / Dm);
  const float var  = rq[0] * (1.0f / Dm) - mean * mean;
  const float inv  = rsqrtf(var + 1e-5f);
  float* orow = out + (size_t)t * Dm;
  for (int i = threadIdx.x; i < Dm; i += 256)
    orow[i] = (xr[i] - mean) * inv * g[i] + bta[i];
}

// ---------------------------------------------------------------------------
// Knowledge gate (in place)
// ---------------------------------------------------------------------------
__global__ __launch_bounds__(256)
void k_gate(float* __restrict__ x, const float* __restrict__ wg,
            const float* __restrict__ bg, const float* __restrict__ gk,
            const float* __restrict__ bk) {
  const int t = blockIdx.x;
  float* xr = x + (size_t)t * Dm;
  __shared__ float rd[256], rs[256], rq[256];
  float d = 0.f, s = 0.f, q = 0.f;
  for (int i = threadIdx.x; i < Dm; i += 256) {
    float v = xr[i];
    d += v * wg[i]; s += v; q += v * v;
  }
  rd[threadIdx.x] = d; rs[threadIdx.x] = s; rq[threadIdx.x] = q;
  __syncthreads();
  for (int off = 128; off > 0; off >>= 1) {
    if (threadIdx.x < off) {
      rd[threadIdx.x] += rd[threadIdx.x + off];
      rs[threadIdx.x] += rs[threadIdx.x + off];
      rq[threadIdx.x] += rq[threadIdx.x + off];
    }
    __syncthreads();
  }
  const float gate = 1.0f / (1.0f + __expf(-(rd[0] + bg[0])));
  const float mean = rs[0] * (1.0f / Dm);
  const float var  = rq[0] * (1.0f / Dm) - mean * mean;
  const float inv  = rsqrtf(var + 1e-5f);
  for (int i = threadIdx.x; i < Dm; i += 256) {
    float v  = xr[i];
    float ln = (v - mean) * inv * gk[i] + bk[i];
    xr[i] = v * gate + ln * (1.0f - gate);
  }
}

// ---------------------------------------------------------------------------
// Tiled bf16-WMMA GEMM: C[M,N] = act(A[M,K] @ W[K,N] + bias) (+ addsrc)
// 256 threads = 8 waves (4x2). Block tile 128x64, wave tile 32x32.
// Double-buffered LDS staging, K-step 64 => 8 WMMAs per stage per wave.
// ---------------------------------------------------------------------------
#define ASTRIDE 72
template <int ACT>   // 0 = none, 1 = exact gelu
__global__ __launch_bounds__(256)
void k_gemm_wmma(const float* __restrict__ A, const float* __restrict__ W,
                 const float* __restrict__ bias, const float* __restrict__ addsrc,
                 float* __restrict__ C, int M, int N, int K) {
  __shared__ __align__(16) bf16 As[2][128 * ASTRIDE];   // 2 x 18 KB
  __shared__ __align__(16) bf16 Ws[2][64 * ASTRIDE];    // 2 x  9 KB

  const int tid  = threadIdx.x;
  const int lane = tid & 31;
  const int wid  = tid >> 5;     // 0..7
  const int wm   = wid & 3;      // 32-row strip
  const int wn   = wid >> 2;     // 32-col strip
  const int bm   = blockIdx.y * 128;
  const int bn   = blockIdx.x * 64;

  v8f acc00 = {}, acc01 = {}, acc10 = {}, acc11 = {};

  // staging decomposition
  const int ra = tid >> 1;              // 0..127  (A rows)
  const int ca = (tid & 1) * 32;        // 0 / 32
  const int rw = tid >> 2;              // 0..63   (W k-rows)
  const int cw = (tid & 3) * 16;        // 0..48

  // ---- prologue: stage slab 0 ----
  {
    const float* sa = A + (size_t)(bm + ra) * K + ca;
    cvt16(sa,      &As[0][ra * ASTRIDE + ca]);
    cvt16(sa + 16, &As[0][ra * ASTRIDE + ca + 16]);
    const float* sw = W + (size_t)rw * N + bn + cw;
    bf16 tmp[16];
    cvt16(sw, tmp);
#pragma unroll
    for (int j = 0; j < 16; ++j) Ws[0][(cw + j) * ASTRIDE + rw] = tmp[j];
  }
  __syncthreads();

  int p = 0;
  for (int k0 = 0; k0 < K; k0 += 64) {
    // ---- stage next slab into buffer p^1 while computing on p ----
    if (k0 + 64 < K) {
      const float* sa = A + (size_t)(bm + ra) * K + k0 + 64 + ca;
      cvt16(sa,      &As[p ^ 1][ra * ASTRIDE + ca]);
      cvt16(sa + 16, &As[p ^ 1][ra * ASTRIDE + ca + 16]);
      __builtin_prefetch(sa + 64, 0, 0);           // global_prefetch_b8
      const float* sw = W + (size_t)(k0 + 64 + rw) * N + bn + cw;
      bf16 tmp[16];
      cvt16(sw, tmp);
#pragma unroll
      for (int j = 0; j < 16; ++j) Ws[p ^ 1][(cw + j) * ASTRIDE + rw] = tmp[j];
    }
    // ---- compute on buffer p: 2 K-steps x (2 A-frags x 2 B-frags) ----
#pragma unroll
    for (int ks = 0; ks < 64; ks += 32) {
      const bf16* abase = &As[p][(wm * 32 + (lane & 15)) * ASTRIDE + ks];
      const bf16* bbase = &Ws[p][(wn * 32 + (lane & 15)) * ASTRIDE + ks];
      FragBF a0 = frag_a_lds(abase, lane);
      FragBF a1 = frag_a_lds(abase + 16 * ASTRIDE, lane);
      FragBF b0 = frag_b_lds(bbase, lane);
      FragBF b1 = frag_b_lds(bbase + 16 * ASTRIDE, lane);
      acc00 = wmma_bf16(a0, b0, acc00);
      acc01 = wmma_bf16(a0, b1, acc01);
      acc10 = wmma_bf16(a1, b0, acc10);
      acc11 = wmma_bf16(a1, b1, acc11);
    }
    __syncthreads();
    p ^= 1;
  }

  // ---- writeback: VGPR r -> row r (lanes 0-15) / row 8+r (lanes 16-31) ----
  const int rbase = bm + wm * 32 + ((lane >> 4) * 8);
  const int c0    = bn + wn * 32 + (lane & 15);
  const bool hasadd = (addsrc != nullptr);
#pragma unroll
  for (int i = 0; i < 2; ++i) {        // A sub-tile (rows +16*i)
#pragma unroll
    for (int j = 0; j < 2; ++j) {      // B sub-tile (cols +16*j)
      const v8f acc = (i == 0) ? (j == 0 ? acc00 : acc01)
                               : (j == 0 ? acc10 : acc11);
      const int col = c0 + j * 16;
      const float bcol = bias[col];
#pragma unroll
      for (int r = 0; r < 8; ++r) {
        const int row = rbase + i * 16 + r;
        float v = acc[r] + bcol;
        if (ACT == 1) v = gelu_exact(v);
        if (hasadd) v += addsrc[(size_t)row * N + col];
        C[(size_t)row * N + col] = v;
      }
    }
  }
}

// ---------------------------------------------------------------------------
// Causal attention, one (b, h, 16-query tile) per block; 128 threads / 4 waves.
// ---------------------------------------------------------------------------
__global__ __launch_bounds__(128)
void k_attn_wmma(const float* __restrict__ qkv, float* __restrict__ y) {
  const int qt = blockIdx.x & 63;
  const int h  = (blockIdx.x >> 6) & (Hm - 1);
  const int b  = blockIdx.x >> 10;
  const int tid = threadIdx.x, lane = tid & 31, wid = tid >> 5;  // wid 0..3

  __shared__ __align__(16) bf16  Qs[16 * 72];     // Q tile [q][d]
  __shared__ __align__(16) bf16  Ss[16 * 1056];   // scores -> probs
  __shared__ __align__(16) bf16  Vs[64 * 40];     // V tile transposed [d][k]
  __shared__ float red[16 * 8];
  __shared__ float rowM[16], rowS[16];

  const size_t tok = 3 * (size_t)Dm;
  const int hd0 = h * 64;

  // ---- phase 1: stage Q tile (16x64) as bf16 ----
  {
    int r = tid >> 3;                 // 0..15
    int c = (tid & 7) * 8;            // 0..56
    const float* qp = qkv + ((size_t)(b * Tm + qt * 16 + r)) * tok + hd0 + c;
    float4 f0 = *(const float4*)qp;
    float4 f1 = *(const float4*)(qp + 4);
    bf16* dst = Qs + r * 72 + c;
    dst[0] = (bf16)f0.x; dst[1] = (bf16)f0.y; dst[2] = (bf16)f0.z; dst[3] = (bf16)f0.w;
    dst[4] = (bf16)f1.x; dst[5] = (bf16)f1.y; dst[6] = (bf16)f1.z; dst[7] = (bf16)f1.w;
  }
  __syncthreads();

  // ---- phase 2: scores via WMMA, causal mask, scale ----
  const int nkt = qt + 1;
  const float scale = 0.125f;                 // 1/sqrt(64)
  for (int kt = wid; kt < nkt; kt += 4) {
    const int key = kt * 16 + (lane & 15);
    const float* kp = qkv + ((size_t)(b * Tm + key)) * tok + Dm + hd0;
    const int kb2 = (lane < 16) ? 0 : 16;
    v8f acc = {};
#pragma unroll
    for (int ks = 0; ks < 2; ++ks) {
      FragBF bfr;
      cvt16(kp + ks * 32 + kb2, bfr.s);
      FragBF a = frag_a_lds(Qs + (lane & 15) * 72 + ks * 32, lane);
      acc = wmma_bf16(a, bfr, acc);
    }
    const int n = lane & 15;
    const int kglob = kt * 16 + n;
#pragma unroll
    for (int r = 0; r < 8; ++r) {
      const int qrow  = r + ((lane >> 4) * 8);
      const int qglob = qt * 16 + qrow;
      float sv = acc[r] * scale;
      if (kglob > qglob) sv = -3.0e38f;
      Ss[qrow * 1056 + kt * 16 + n] = (bf16)sv;
    }
  }
  __syncthreads();

  // ---- phase 3: softmax per row; scores -> probs in place ----
  const int ncols   = nkt * 16;
  const int ncols32 = (ncols + 31) & ~31;
  const int row = tid >> 3, sub = tid & 7;
  float mx = -3.4e38f;
  for (int c = sub; c < ncols; c += 8) mx = fmaxf(mx, (float)Ss[row * 1056 + c]);
  red[row * 8 + sub] = mx;
  __syncthreads();
  if (sub == 0) {
    float mm = red[row * 8];
#pragma unroll
    for (int i = 1; i < 8; ++i) mm = fmaxf(mm, red[row * 8 + i]);
    rowM[row] = mm;
  }
  __syncthreads();
  const float mm = rowM[row];
  float se = 0.f;
  for (int c = sub; c < ncols; c += 8) se += __expf((float)Ss[row * 1056 + c] - mm);
  __syncthreads();
  red[row * 8 + sub] = se;
  __syncthreads();
  if (sub == 0) {
    float ss = 0.f;
#pragma unroll
    for (int i = 0; i < 8; ++i) ss += red[row * 8 + i];
    rowS[row] = ss;
  }
  __syncthreads();
  const float inv = 1.0f / rowS[row];
  for (int c = sub; c < ncols32; c += 8) {
    float p = (c < ncols) ? __expf((float)Ss[row * 1056 + c] - mm) * inv : 0.0f;
    Ss[row * 1056 + c] = (bf16)p;
  }

  // ---- phase 4: Y[16x64] = P @ V; each wave owns one 16-feature slice ----
  v8f yacc = {};
  const int nkt2 = ncols32 >> 5;
  for (int kt2 = 0; kt2 < nkt2; ++kt2) {
    {
      const int key = kt2 * 32 + (tid >> 2);
      const int f0  = (tid & 3) * 16;
      const float* vp = qkv + ((size_t)(b * Tm + key)) * tok + 2 * Dm + hd0 + f0;
      bf16 tmp[16];
      cvt16(vp, tmp);
      const int kk = tid >> 2;
#pragma unroll
      for (int j = 0; j < 16; ++j) Vs[(f0 + j) * 40 + kk] = tmp[j];
    }
    __syncthreads();
    FragBF a  = frag_a_lds(Ss + (lane & 15) * 1056 + kt2 * 32, lane);
    FragBF bv = frag_b_lds(Vs + (wid * 16 + (lane & 15)) * 40, lane);
    yacc = wmma_bf16(a, bv, yacc);
    __syncthreads();
  }

  // ---- writeback ----
  const int n = lane & 15;
#pragma unroll
  for (int r = 0; r < 8; ++r) {
    const int qrow = r + ((lane >> 4) * 8);
    const int t = qt * 16 + qrow;
    y[((size_t)(b * Tm + t)) * Dm + hd0 + wid * 16 + n] = yacc[r];
  }
}

// ---------------------------------------------------------------------------
// Host launcher
// ---------------------------------------------------------------------------
extern "C" void kernel_launch(void* const* d_in, const int* in_sizes, int n_in,
                              void* d_out, int out_size, void* d_ws, size_t ws_size,
                              hipStream_t stream) {
  const int*   idx     = (const int*)  d_in[0];
  const float* tok_emb = (const float*)d_in[1];
  const float* pos_emb = (const float*)d_in[2];
  const float* ln1_g   = (const float*)d_in[3];
  const float* ln1_b   = (const float*)d_in[4];
  const float* Wqkv    = (const float*)d_in[5];
  const float* bqkv    = (const float*)d_in[6];
  const float* Wproj   = (const float*)d_in[7];
  const float* bproj   = (const float*)d_in[8];
  const float* Wgate   = (const float*)d_in[9];
  const float* bgate   = (const float*)d_in[10];
  const float* lnk_g   = (const float*)d_in[11];
  const float* lnk_b   = (const float*)d_in[12];
  const float* ln2_g   = (const float*)d_in[13];
  const float* ln2_b   = (const float*)d_in[14];
  const float* Wfc     = (const float*)d_in[15];
  const float* bfc     = (const float*)d_in[16];
  const float* Wp      = (const float*)d_in[17];
  const float* bp      = (const float*)d_in[18];
  const float* lnf_g   = (const float*)d_in[19];
  const float* lnf_b   = (const float*)d_in[20];
  const float* Whead   = (const float*)d_in[21];
  const float* bhead   = (const float*)d_in[22];

  float* x    = (float*)d_ws;
  float* hbuf = x    + (size_t)Mm * Dm;
  float* qkv  = hbuf + (size_t)Mm * Dm;
  float* ybuf = qkv  + (size_t)Mm * 3 * Dm;
  float* mbuf = ybuf + (size_t)Mm * Dm;

  k_embed<<<Mm, 256, 0, stream>>>(idx, tok_emb, pos_emb, x);

  for (int l = 0; l < Lm; ++l) {
    const size_t lD = (size_t)l * Dm;
    k_layernorm<<<Mm, 256, 0, stream>>>(x, ln1_g + lD, ln1_b + lD, hbuf);
    k_gemm_wmma<0><<<dim3(3 * Dm / 64, Mm / 128), 256, 0, stream>>>(
        hbuf, Wqkv + (size_t)l * Dm * 3 * Dm, bqkv + (size_t)l * 3 * Dm,
        nullptr, qkv, Mm, 3 * Dm, Dm);
    k_attn_wmma<<<Bm * Hm * (Tm / 16), 128, 0, stream>>>(qkv, ybuf);
    k_gemm_wmma<0><<<dim3(Dm / 64, Mm / 128), 256, 0, stream>>>(
        ybuf, Wproj + (size_t)l * Dm * Dm, bproj + lD, x, x, Mm, Dm, Dm);
    k_gate<<<Mm, 256, 0, stream>>>(x, Wgate + lD, bgate + l, lnk_g + lD, lnk_b + lD);
    k_layernorm<<<Mm, 256, 0, stream>>>(x, ln2_g + lD, ln2_b + lD, hbuf);
    k_gemm_wmma<1><<<dim3(4 * Dm / 64, Mm / 128), 256, 0, stream>>>(
        hbuf, Wfc + (size_t)l * Dm * 4 * Dm, bfc + (size_t)l * 4 * Dm,
        nullptr, mbuf, Mm, 4 * Dm, Dm);
    k_gemm_wmma<0><<<dim3(Dm / 64, Mm / 128), 256, 0, stream>>>(
        mbuf, Wp + (size_t)l * 4 * Dm * Dm, bp + lD, x, x, Mm, Dm, 4 * Dm);
  }

  k_layernorm<<<Mm, 256, 0, stream>>>(x, lnf_g, lnf_b, hbuf);
  k_gemm_wmma<0><<<dim3(Vm / 64, Mm / 128), 256, 0, stream>>>(
      hbuf, Whead, bhead, nullptr, (float*)d_out, Mm, Vm, Dm);
}
